// SimpleGCN_31301721653263
// MI455X (gfx1250) — matrix operand order repaired and checked by reading, not compiled, not run on hardware
//
#include <hip/hip_runtime.h>
#include <hip/hip_bf16.h>

typedef long long i64;
typedef __attribute__((ext_vector_type(16))) __bf16 v16bf;
typedef __attribute__((ext_vector_type(8)))  float  v8f;

#define HD 128          // hidden / in_channels
#define TILE_M 16       // node rows per block

// ---------------- utility kernels ----------------

__global__ void gcn_zero_f32(float* __restrict__ p, int n) {
    int i = blockIdx.x * blockDim.x + threadIdx.x;
    if (i < n) p[i] = 0.0f;
}

// Pack W (f32, [K=128][N=128] row-major) into WMMA B-fragment-major bf16:
// slot t = ((ct*4 + ks)*32 + lane)*16 + j  holds  W[k][n] with
//   k = ks*32 + (lane>>4)*16 + j,  n = ct*16 + (lane&15)
// so each lane's 16 B elements are 32 contiguous, 32B-aligned bytes.
__global__ void gcn_pack_w_bf16(const float* __restrict__ W, __bf16* __restrict__ Wp) {
    int t = blockIdx.x * blockDim.x + threadIdx.x;
    if (t >= HD * HD) return;
    int j    = t & 15;
    int lane = (t >> 4) & 31;
    int ks   = (t >> 9) & 3;
    int ct   = t >> 11;
    int k = ks * 32 + (lane >> 4) * 16 + j;
    int n = ct * 16 + (lane & 15);
    Wp[t] = (__bf16)W[k * HD + n];
}

__global__ void gcn_deg_count(const i64* __restrict__ dst, float* __restrict__ deg, int e) {
    int i = blockIdx.x * blockDim.x + threadIdx.x;
    if (i < e) atomicAdd(&deg[(int)dst[i]], 1.0f);
}

__global__ void gcn_deg_finalize(float* __restrict__ deg, int n) {
    int i = blockIdx.x * blockDim.x + threadIdx.x;
    if (i < n) deg[i] = rsqrtf(deg[i] + 1.0f);   // +1 = self loop; overwrite with deg^-1/2
}

// ---------------- WMMA GEMM + self-loop epilogue ----------------
// T[n,128] = X[n,:] @ W      (bf16 A/B, f32 accumulate)
// P[n,128] = T[n,:] * dinv[n]^2   (self-loop term == aggregation init)

__global__ void gcn_gemm_wmma(const float* __restrict__ X,
                              const __bf16* __restrict__ Wp,
                              const float* __restrict__ dinv,
                              float* __restrict__ T,
                              float* __restrict__ P, int nrows) {
    __shared__ __bf16 ldsA[TILE_M][HD];

    const int base = blockIdx.x * TILE_M;

    // cooperative load of 16x128 f32 A-tile, convert to bf16 in LDS
    for (int i = threadIdx.x; i < TILE_M * HD; i += blockDim.x) {
        int r = i >> 7, c = i & 127;
        float v = (base + r < nrows) ? X[(i64)(base + r) * HD + c] : 0.0f;
        ldsA[r][c] = (__bf16)v;
    }
    __syncthreads();

    const int wave = threadIdx.x >> 5;      // 8 waves -> 8 column tiles of 16
    const int lane = threadIdx.x & 31;
    const int half = lane >> 4;             // wave32 half
    const int l15  = lane & 15;

    v8f acc = {};
    #pragma unroll
    for (int ks = 0; ks < 4; ++ks) {
        // A 16x32 bf16 fragment: one contiguous 32B run per lane (k = half*16 + j), matching B
        v16bf a;
        #pragma unroll
        for (int j = 0; j < 16; ++j) {
            int ka = ks * 32 + half * 16 + j;
            a[j] = ldsA[l15][ka];
        }
        // B fragment: 32 contiguous bytes from the pre-swizzled weight slab
        const v16bf b = *reinterpret_cast<const v16bf*>(
            Wp + (((size_t)wave * 4 + (size_t)ks) * 32 + (size_t)lane) * 16);
        acc = __builtin_amdgcn_wmma_f32_16x16x32_bf16(
            /*neg_a=*/false, a, /*neg_b=*/false, b,
            /*c_mod=*/(short)0, acc, /*reuse_a=*/false, /*reuse_b=*/false);
    }

    // D layout: VGPR r holds row M = r + 8*half, col = l15
    const int col = wave * 16 + l15;
    if (base + TILE_M <= nrows) {
        #pragma unroll
        for (int r = 0; r < 8; ++r) {
            int m = base + r + half * 8;
            float v  = acc[r];
            float di = dinv[m];
            T[(i64)m * HD + col] = v;
            P[(i64)m * HD + col] = v * di * di;
        }
    } else {
        #pragma unroll
        for (int r = 0; r < 8; ++r) {
            int m = base + r + half * 8;
            if (m < nrows) {
                float v  = acc[r];
                float di = dinv[m];
                T[(i64)m * HD + col] = v;
                P[(i64)m * HD + col] = v * di * di;
            }
        }
    }
}

// ---------------- aggregation ----------------

// one wave per edge: float4 gather of h[src], scaled atomic scatter into h[dst]
__global__ void gcn_edge_scatter(const i64* __restrict__ src, const i64* __restrict__ dst,
                                 const float* __restrict__ dinv, const float* __restrict__ T,
                                 float* __restrict__ O, int e) {
    int eid = blockIdx.x * 8 + (threadIdx.x >> 5);
    if (eid >= e) return;
    int lane = threadIdx.x & 31;
    int s = (int)src[eid];
    int d = (int)dst[eid];
    float norm = dinv[s] * dinv[d];
    const float4 v = *(const float4*)(T + (i64)s * HD + lane * 4);
    float* o = O + (i64)d * HD + lane * 4;
    atomicAdd(o + 0, v.x * norm);
    atomicAdd(o + 1, v.y * norm);
    atomicAdd(o + 2, v.z * norm);
    atomicAdd(o + 3, v.w * norm);
}

__global__ void gcn_bias_relu(float* __restrict__ O, const float* __restrict__ b, int n) {
    int t = blockIdx.x * blockDim.x + threadIdx.x;
    if (t < n * HD) {
        float v = O[t] + b[t & 127];
        O[t] = v > 0.0f ? v : 0.0f;
    }
}

// ---------------- pooling + classifier ----------------

__global__ void gcn_pool_cnt(const i64* __restrict__ batch, float* __restrict__ cnt, int n) {
    int i = blockIdx.x * blockDim.x + threadIdx.x;
    if (i < n) atomicAdd(&cnt[(int)batch[i]], 1.0f);
}

__global__ void gcn_pool_add(const i64* __restrict__ batch, const float* __restrict__ h,
                             float* __restrict__ pooled, int n) {
    int t = blockIdx.x * blockDim.x + threadIdx.x;
    if (t < n * HD) {
        int node = t >> 7;
        int f = t & 127;
        int g = (int)batch[node];
        atomicAdd(&pooled[g * HD + f], h[t]);
    }
}

__global__ void gcn_final_linear(const float* __restrict__ pooled, const float* __restrict__ cnt,
                                 const float* __restrict__ Wl, const float* __restrict__ bl,
                                 float* __restrict__ out, int g) {
    int t = blockIdx.x * blockDim.x + threadIdx.x;
    if (t < g * 2) {
        int gi = t >> 1, c = t & 1;
        float inv = 1.0f / fmaxf(cnt[gi], 1.0f);
        float acc = bl[c];
        for (int f = 0; f < HD; ++f)
            acc += pooled[gi * HD + f] * inv * Wl[f * 2 + c];
        out[t] = acc;
    }
}

// ---------------- launcher ----------------

extern "C" void kernel_launch(void* const* d_in, const int* in_sizes, int n_in,
                              void* d_out, int out_size, void* d_ws, size_t ws_size,
                              hipStream_t stream) {
    const float* x    = (const float*)d_in[0];
    const i64*  ei    = (const i64*)d_in[1];
    const i64*  batch = (const i64*)d_in[2];
    const float* W1   = (const float*)d_in[3];
    const float* b1   = (const float*)d_in[4];
    const float* W2   = (const float*)d_in[5];
    const float* b2   = (const float*)d_in[6];
    const float* W3   = (const float*)d_in[7];
    const float* b3   = (const float*)d_in[8];
    const float* Wl   = (const float*)d_in[9];
    const float* bl   = (const float*)d_in[10];
    float* out = (float*)d_out;

    const int N = in_sizes[2];          // batch has N elements
    const int E = in_sizes[1] / 2;      // edge_index is [2, E]
    const int G = out_size / 2;
    const i64* src = ei;
    const i64* dst = ei + E;

    // carve workspace (256B aligned)
    char* ws = (char*)d_ws;
    size_t off = 0;
    auto alloc = [&](size_t bytes) -> void* {
        off = (off + 255) & ~(size_t)255;
        void* p = ws + off;
        off += bytes;
        return p;
    };
    __bf16* Wp     = (__bf16*)alloc((size_t)3 * HD * HD * sizeof(__bf16));
    float*  dinv   = (float*) alloc((size_t)N * sizeof(float));
    float*  T      = (float*) alloc((size_t)N * HD * sizeof(float));
    float*  P      = (float*) alloc((size_t)N * HD * sizeof(float));
    float*  pooled = (float*) alloc((size_t)G * HD * sizeof(float));
    float*  cnt    = (float*) alloc((size_t)G * sizeof(float));

    const int B = 256;
    const int nhBlocks = (N * HD + B - 1) / B;
    const int rowTiles = (N + TILE_M - 1) / TILE_M;
    const int wBlocks  = (HD * HD + B - 1) / B;

    // weights -> fragment-major bf16 (deterministic, recomputed each launch)
    gcn_pack_w_bf16<<<wBlocks, B, 0, stream>>>(W1, Wp);
    gcn_pack_w_bf16<<<wBlocks, B, 0, stream>>>(W2, Wp + HD * HD);
    gcn_pack_w_bf16<<<wBlocks, B, 0, stream>>>(W3, Wp + 2 * HD * HD);

    // symmetric normalization coefficients
    gcn_zero_f32<<<(N + B - 1) / B, B, 0, stream>>>(dinv, N);
    gcn_deg_count<<<(E + B - 1) / B, B, 0, stream>>>(dst, dinv, E);
    gcn_deg_finalize<<<(N + B - 1) / B, B, 0, stream>>>(dinv, N);

    // three GCN layers; GEMM writes T (=h) and P (=h*dinv^2, aggregation init)
    const float* in = x;
    const float* biases[3] = { b1, b2, b3 };
    for (int l = 0; l < 3; ++l) {
        gcn_gemm_wmma<<<rowTiles, 256, 0, stream>>>(in, Wp + (size_t)l * HD * HD, dinv, T, P, N);
        gcn_edge_scatter<<<(E + 7) / 8, 256, 0, stream>>>(src, dst, dinv, T, P, E);
        gcn_bias_relu<<<nhBlocks, B, 0, stream>>>(P, biases[l], N);
        in = P;
    }

    // global mean pool + classifier
    gcn_zero_f32<<<(G * HD + B - 1) / B, B, 0, stream>>>(pooled, G * HD);
    gcn_zero_f32<<<1, B, 0, stream>>>(cnt, G);
    gcn_pool_cnt<<<(N + B - 1) / B, B, 0, stream>>>(batch, cnt, N);
    gcn_pool_add<<<nhBlocks, B, 0, stream>>>(batch, P, pooled, N);
    gcn_final_linear<<<(G * 2 + 127) / 128, 128, 0, stream>>>(pooled, cnt, Wl, bl, out, G);
}
